// LocalTransformerBlock_24567212933637
// MI455X (gfx1250) — matrix operand order repaired and checked by reading, not compiled, not run on hardware
//
#include <hip/hip_runtime.h>
#include <cstdint>
#include <cstddef>

#define DIMX 1024
#define HEADS 8
#define DH 64
#define INNER 512
#define NQKV 1536
#define SEQ 8192
#define WSZ 512
#define NWIN 16
#define HID 2730
#define HID_P 2752
#define N1V 5460
#define N1P 5504
#define QK_SCALE_F 8.0f
#define SCALE_BASE_F 256.0f

typedef __attribute__((ext_vector_type(16))) __bf16 v16bf;
typedef __attribute__((ext_vector_type(8)))  float  v8f;

static __device__ __forceinline__ v8f wmma_bf16(v16bf a, v16bf b, v8f c) {
  // D = A(16x32 bf16) * B(32x16 bf16) + C(16x16 f32)
  return __builtin_amdgcn_wmma_f32_16x16x32_bf16(false, a, false, b, (short)0, c,
                                                 false, false);
}

// --------------------------- async copy helpers ----------------------------
// GLOBAL_LOAD_ASYNC_TO_LDS_B128: per-lane 16B global->LDS, tracked by ASYNCcnt.
// INST_OFFSET is added to BOTH the LDS and the global address (ISA 15.18.3),
// so one (lds,vaddr) pair serves consecutive 16B beats.
static __device__ __forceinline__ uint32_t lds_off(const void* p) {
  // LDS flat addresses keep the byte offset in bits[31:0].
  return (uint32_t)(uintptr_t)p;
}
static __device__ __forceinline__ void async_copy32(uint32_t lds, const void* g) {
  asm volatile("global_load_async_to_lds_b128 %0, %1, off\n\t"
               "global_load_async_to_lds_b128 %0, %1, off offset:16"
               :: "v"(lds), "v"(g) : "memory");
}
static __device__ __forceinline__ void async_copy64(uint32_t lds, const void* g) {
  asm volatile("global_load_async_to_lds_b128 %0, %1, off\n\t"
               "global_load_async_to_lds_b128 %0, %1, off offset:16\n\t"
               "global_load_async_to_lds_b128 %0, %1, off offset:32\n\t"
               "global_load_async_to_lds_b128 %0, %1, off offset:48"
               :: "v"(lds), "v"(g) : "memory");
}
static __device__ __forceinline__ void async_wait0() {
  asm volatile("s_wait_asynccnt 0x0" ::: "memory");
}

// --------------------------- WMMA fragment loads ---------------------------
// A fragment (16x32, MxK): lane = 16*khalf + m; element e -> K = khalf*16 + e.
static __device__ __forceinline__ v16bf load_a_frag(const __bf16* base, int stride, int lane) {
  const int m = lane & 15, hl = lane >> 4;
  const __bf16* p = base + m * stride;
  v16bf a;
#pragma unroll
  for (int e = 0; e < 16; ++e) {
    int k = hl * 16 + e;
    a[e] = p[k];
  }
  return a;
}
// B fragment from an N-major (reduction-dim-contiguous) tile:
// lane = 16*khalf + n, element e -> K = khalf*16 + e => 16 contiguous elems.
static __device__ __forceinline__ v16bf load_bt_frag(const __bf16* base, int stride, int lane) {
  const int n = lane & 15, hl = lane >> 4;
  const __bf16* p = base + n * stride + hl * 16;
  v16bf b;
#pragma unroll
  for (int e = 0; e < 16; ++e) b[e] = p[e];
  return b;
}

// ---------------------------------------------------------------------------
// bf16 WMMA GEMM: C(MxN) = A(MxK) * Bt(NxK)^T [+bias] [+resid]
// BLOCK 128x128, K-step 32, 256 threads = 8 waves (2x4), wave tile 64x32.
// Double-buffered LDS, tiles staged with GLOBAL_LOAD_ASYNC_TO_LDS_B128,
// one s_wait_asynccnt + one barrier per k-tile.
// M % 128 == 0, N % 128 == 0, K % 32 == 0 (caller pads buffers).
// ---------------------------------------------------------------------------
template <bool OUT_BF16>
__global__ __launch_bounds__(256) void gemm_wmma(
    const __bf16* __restrict__ A, int lda,
    const __bf16* __restrict__ Bt, int ldb,   // N x K row-major
    void* __restrict__ Cv, int ldc,
    const float* __restrict__ bias, int nvalid,
    const float* __restrict__ resid,
    int M, int N, int K) {
  __shared__ __align__(16) __bf16 sA[2][128][40];   // 80B row stride
  __shared__ __align__(16) __bf16 sB[2][128][40];   // transposed tile: [n][k]

  const int t = threadIdx.x;
  const int lane = t & 31, wid = t >> 5;
  const int wm = wid >> 2, wn = wid & 3;
  const int gm = blockIdx.y * 128, gn = blockIdx.x * 128;

  v8f acc[4][2];
#pragma unroll
  for (int i = 0; i < 4; ++i)
#pragma unroll
    for (int j = 0; j < 2; ++j)
#pragma unroll
      for (int r = 0; r < 8; ++r) acc[i][j][r] = 0.f;

  const int srow = t >> 1, scol = (t & 1) * 16;  // 128 rows x 2 halves of 16

  auto issue_tile = [&](int k0, int buf) {
    async_copy32(lds_off(&sA[buf][srow][scol]),
                 A + (size_t)(gm + srow) * lda + k0 + scol);
    async_copy32(lds_off(&sB[buf][srow][scol]),
                 Bt + (size_t)(gn + srow) * ldb + k0 + scol);
  };

  const int nt = K >> 5;
  issue_tile(0, 0);
  async_wait0();
  __syncthreads();

  for (int it = 0; it < nt; ++it) {
    if (it + 1 < nt) issue_tile((it + 1) << 5, (it + 1) & 1);
    const int buf = it & 1;

    v16bf bf[2];
#pragma unroll
    for (int j = 0; j < 2; ++j)
      bf[j] = load_bt_frag(&sB[buf][wn * 32 + j * 16][0], 40, lane);
#pragma unroll
    for (int i = 0; i < 4; ++i) {
      v16bf af = load_a_frag(&sA[buf][wm * 64 + i * 16][0], 40, lane);
#pragma unroll
      for (int j = 0; j < 2; ++j) acc[i][j] = wmma_bf16(af, bf[j], acc[i][j]);
    }
    async_wait0();     // next tile has landed (issued before compute)
    __syncthreads();   // all waves done reading buf before it is re-written
  }

  const int n_in = lane & 15, hl = lane >> 4;
#pragma unroll
  for (int i = 0; i < 4; ++i) {
#pragma unroll
    for (int j = 0; j < 2; ++j) {
#pragma unroll
      for (int r = 0; r < 8; ++r) {
        const int row = gm + wm * 64 + i * 16 + r + 8 * hl;
        const int col = gn + wn * 32 + j * 16 + n_in;
        float v = acc[i][j][r];
        if (bias != nullptr && col < nvalid) v += bias[col];
        if (resid != nullptr) v += resid[(size_t)row * ldc + col];
        if (OUT_BF16)
          ((__bf16*)Cv)[(size_t)row * ldc + col] = (__bf16)v;
        else
          ((float*)Cv)[(size_t)row * ldc + col] = v;
      }
    }
  }
  (void)M; (void)N;
}

// ---------------------------------------------------------------------------
// LayerNorm(fp32 in) -> bf16 out. One block per row of 1024.
// ---------------------------------------------------------------------------
__global__ __launch_bounds__(256) void ln_bf16_kernel(
    const float* __restrict__ x, const float* __restrict__ g,
    const float* __restrict__ b, __bf16* __restrict__ out) {
  __shared__ float red[256];
  const int row = blockIdx.x, t = threadIdx.x;
  const float* xr = x + (size_t)row * DIMX;
  float v[4];
  float s = 0.f;
#pragma unroll
  for (int i = 0; i < 4; ++i) { v[i] = xr[t * 4 + i]; s += v[i]; }
  red[t] = s; __syncthreads();
  for (int o = 128; o > 0; o >>= 1) { if (t < o) red[t] += red[t + o]; __syncthreads(); }
  const float mu = red[0] * (1.0f / DIMX);
  __syncthreads();
  float s2 = 0.f;
#pragma unroll
  for (int i = 0; i < 4; ++i) { float d = v[i] - mu; s2 += d * d; }
  red[t] = s2; __syncthreads();
  for (int o = 128; o > 0; o >>= 1) { if (t < o) red[t] += red[t + o]; __syncthreads(); }
  const float inv = rsqrtf(red[0] * (1.0f / DIMX) + 1e-5f);
#pragma unroll
  for (int i = 0; i < 4; ++i) {
    const int c = t * 4 + i;
    out[(size_t)row * DIMX + c] = (__bf16)((v[i] - mu) * inv * g[c] + b[c]);
  }
}

// ---------------------------------------------------------------------------
// QKV post: l2norm per head, q/k scales, xpos rotary, fold QK_SCALE into q.
// q/k written [head][token][64]; v written TRANSPOSED [head][dim][token] so
// attention can stage V with reduction-dim(keys)-contiguous async copies.
// ---------------------------------------------------------------------------
__global__ __launch_bounds__(64) void qkv_post_kernel(
    const __bf16* __restrict__ qkv, const float* __restrict__ q_scale,
    const float* __restrict__ k_scale, __bf16* __restrict__ qo,
    __bf16* __restrict__ ko, __bf16* __restrict__ vt) {
  const int tok = blockIdx.x, h = blockIdx.y, d = threadIdx.x;
  const __bf16* row = qkv + (size_t)tok * NQKV + h * DH;
  float q = (float)row[d];
  float k = (float)row[INNER + d];
  const float v = (float)row[2 * INNER + d];

  __shared__ float sq[DH], sk[DH];
  sq[d] = q * q; sk[d] = k * k;
  __syncthreads();
  float qs = 0.f, ks = 0.f;
  for (int i = 0; i < DH; ++i) { qs += sq[i]; ks += sk[i]; }
  q = q / fmaxf(sqrtf(qs), 1e-12f) * q_scale[d];
  k = k / fmaxf(sqrtf(ks), 1e-12f) * k_scale[d];
  __syncthreads();
  sq[d] = q; sk[d] = k;
  __syncthreads();
  const float qr = (d < 32) ? -sq[d + 32] : sq[d - 32];   // rotate_half
  const float kr = (d < 32) ? -sk[d + 32] : sk[d - 32];

  const int j = d & 31;
  const float inv_freq = __powf(10000.f, -(float)(2 * j) / (float)DH);
  const float fr = (float)tok * inv_freq;
  const float c = __cosf(fr), sn = __sinf(fr);
  const float sv = ((float)(2 * j) + 0.4f * DH) / (1.4f * DH);
  const float power = ((float)tok - (float)(SEQ / 2)) / SCALE_BASE_F;
  const float scale = __powf(sv, power);
  const float qf = (q * c + qr * sn) * scale * QK_SCALE_F;
  const float kf = (k * c + kr * sn) / scale;

  const size_t o = ((size_t)h * SEQ + tok) * DH + d;
  qo[o] = (__bf16)qf; ko[o] = (__bf16)kf;
  vt[((size_t)h * DH + d) * SEQ + tok] = (__bf16)v;
}

// ---------------------------------------------------------------------------
// Local windowed attention, flash-style online softmax, WMMA for S and P*V.
// grid = (rowblocks=8, windows=16, heads=8), block = 128 (4 waves x 16 q-rows).
// K/V streamed with async-to-LDS double buffering: 1 barrier per 32-key chunk.
// V kept key-contiguous ([dim][key] in LDS) so P*V B-frags are 2x ds_load_b128.
// Wave-uniform chunk classification: fully-masked chunks skip all compute
// (~half the chunks), fully-visible chunks skip mask VALU.
// ---------------------------------------------------------------------------
__global__ __launch_bounds__(128) void attn_kernel(
    const __bf16* __restrict__ qb, const __bf16* __restrict__ kb,
    const __bf16* __restrict__ vt, __bf16* __restrict__ ob) {
  __shared__ __align__(16) __bf16 sQ[64][72];
  __shared__ __align__(16) __bf16 sK[2][32][72];   // [key][dim]
  __shared__ __align__(16) __bf16 sVt[2][64][40];  // [dim][key]
  __shared__ __align__(16) __bf16 sP[4][16][32];

  const int rb = blockIdx.x, w = blockIdx.y, h = blockIdx.z;
  const int t = threadIdx.x, lane = t & 31, wv = t >> 5;
  const int n_in = lane & 15, hl = lane >> 4;
  const int li_base = rb * 64 + wv * 16;

  // stage Q tile (64 rows x 64 dims) asynchronously
  {
    const int row = t >> 1, cs = (t & 1) * 32;
    async_copy64(lds_off(&sQ[row][cs]),
                 qb + ((size_t)h * SEQ + (size_t)w * WSZ + rb * 64 + row) * DH + cs);
  }

  const int krow = t >> 2, kcs = (t & 3) * 16;   // K: 32 keys x 64 dims
  const int vrow = t >> 1, vcs = (t & 1) * 16;   // V: 64 dims x 32 keys
  auto issue_kv = [&](int kc, int buf) {
    const int jj = kc * 32 + krow;
    const int pos = (jj < WSZ) ? ((w - 1) * WSZ + jj) : (w * WSZ + (jj - WSZ));
    async_copy32(lds_off(&sK[buf][krow][kcs]),
                 kb + ((size_t)h * SEQ + pos) * DH + kcs);
    const int pos0 = (kc < 16) ? ((w - 1) * WSZ + kc * 32) : (w * WSZ + (kc - 16) * 32);
    async_copy32(lds_off(&sVt[buf][vrow][vcs]),
                 vt + ((size_t)h * DH + vrow) * SEQ + pos0 + vcs);
  };

  const int kc0 = (w == 0) ? 16 : 0;  // first window has no previous window
  issue_kv(kc0, 0);
  async_wait0();
  __syncthreads();

  v16bf qa[2];
#pragma unroll
  for (int ka = 0; ka < 2; ++ka) qa[ka] = load_a_frag(&sQ[wv * 16][ka * 32], 72, lane);

  float m_run[8], l_run[8];
  v8f o[4];
#pragma unroll
  for (int r = 0; r < 8; ++r) { m_run[r] = -1e30f; l_run[r] = 0.f; }
#pragma unroll
  for (int f = 0; f < 4; ++f)
#pragma unroll
    for (int r = 0; r < 8; ++r) o[f][r] = 0.f;

  for (int kc = kc0; kc < 32; ++kc) {
    const int buf = (kc - kc0) & 1;
    if (kc + 1 < 32) issue_kv(kc + 1, buf ^ 1);   // keep pipeline primed (all waves)

    // Wave-uniform chunk classification. Prev-window key c visible iff c>=li;
    // current-window key c visible iff c<=li; li in [li_base, li_base+15].
    const bool is_prev = (kc < 16);
    const int c0 = (is_prev ? kc : (kc - 16)) * 32;
    const bool full_mask = is_prev ? (c0 + 31 < li_base) : (c0 > li_base + 15);
    const bool full_vis  = is_prev ? (c0 >= li_base + 15) : (c0 + 31 <= li_base);

    if (!full_mask) {
      // S chunk = Q(16x64) * K^T(64x32): two 16x16 C frags, 2 WMMA each.
      v8f s0, s1;
#pragma unroll
      for (int r = 0; r < 8; ++r) { s0[r] = 0.f; s1[r] = 0.f; }
#pragma unroll
      for (int ka = 0; ka < 2; ++ka) {
        v16bf bk0, bk1;
#pragma unroll
        for (int e = 0; e < 16; ++e) {
          bk0[e] = sK[buf][n_in][ka * 32 + hl * 16 + e];
          bk1[e] = sK[buf][16 + n_in][ka * 32 + hl * 16 + e];
        }
        s0 = wmma_bf16(qa[ka], bk0, s0);
        s1 = wmma_bf16(qa[ka], bk1, s1);
      }

      // Online softmax (rows live in 16-lane groups; xor 1/2/4/8 stays in-group).
      float p0a[8], p1a[8];
#pragma unroll
      for (int r = 0; r < 8; ++r) {
        const int li = li_base + r + 8 * hl;
        float a0, a1;
        bool vis0 = true, vis1 = true;
        if (full_vis) {
          a0 = s0[r]; a1 = s1[r];
        } else {
          const int cc0 = c0 + n_in;
          const int cc1 = cc0 + 16;
          vis0 = is_prev ? (cc0 >= li) : (cc0 <= li);
          vis1 = is_prev ? (cc1 >= li) : (cc1 <= li);
          a0 = vis0 ? s0[r] : -1e30f;
          a1 = vis1 ? s1[r] : -1e30f;
        }
        float cm = fmaxf(a0, a1);
        cm = fmaxf(cm, __shfl_xor(cm, 1));
        cm = fmaxf(cm, __shfl_xor(cm, 2));
        cm = fmaxf(cm, __shfl_xor(cm, 4));
        cm = fmaxf(cm, __shfl_xor(cm, 8));
        const float mnew = fmaxf(m_run[r], cm);
        const float alpha = __expf(m_run[r] - mnew);
        const float p0 = vis0 ? __expf(a0 - mnew) : 0.f;
        const float p1 = vis1 ? __expf(a1 - mnew) : 0.f;
        float ps = p0 + p1;
        ps += __shfl_xor(ps, 1);
        ps += __shfl_xor(ps, 2);
        ps += __shfl_xor(ps, 4);
        ps += __shfl_xor(ps, 8);
        l_run[r] = l_run[r] * alpha + ps;
        m_run[r] = mnew;
#pragma unroll
        for (int f = 0; f < 4; ++f) o[f][r] *= alpha;
        p0a[r] = p0; p1a[r] = p1;
      }

      // Relayout P (C-layout) -> A fragment via per-wave LDS staging.
      // Same-wave LDS ops are in-order in HW; wave_barrier fences the compiler.
#pragma unroll
      for (int r = 0; r < 8; ++r) {
        const int m = r + 8 * hl;
        sP[wv][m][n_in] = (__bf16)p0a[r];
        sP[wv][m][16 + n_in] = (__bf16)p1a[r];
      }
      __builtin_amdgcn_wave_barrier();
      const v16bf pa = load_a_frag(&sP[wv][0][0], 32, lane);
#pragma unroll
      for (int f = 0; f < 4; ++f) {
        const v16bf bv = load_bt_frag(&sVt[buf][f * 16][0], 40, lane);
        o[f] = wmma_bf16(pa, bv, o[f]);
      }
    }

    async_wait0();     // next chunk landed (issued before compute)
    __syncthreads();   // all waves done reading buf before it is re-written
  }

  // Normalize and write token-major bf16 output [token][h*64+d].
#pragma unroll
  for (int r = 0; r < 8; ++r) {
    const int li = li_base + r + 8 * hl;
    const float invl = 1.0f / fmaxf(l_run[r], 1e-20f);
    const size_t tokoff = (size_t)((size_t)w * WSZ + li) * (HEADS * DH);
#pragma unroll
    for (int f = 0; f < 4; ++f)
      ob[tokoff + h * DH + f * 16 + n_in] = (__bf16)(o[f][r] * invl);
  }
}

// ---------------------------------------------------------------------------
// GEGLU: act[:, 0:2730] = a * gelu_exact(gate); padded to 2752 with zeros.
// ---------------------------------------------------------------------------
__global__ __launch_bounds__(256) void geglu_kernel(const __bf16* __restrict__ u,
                                                    __bf16* __restrict__ act) {
  const size_t total = (size_t)SEQ * HID_P;
  for (size_t i = (size_t)blockIdx.x * blockDim.x + threadIdx.x; i < total;
       i += (size_t)gridDim.x * blockDim.x) {
    const size_t r = i / HID_P;
    const int c = (int)(i % HID_P);
    float v = 0.f;
    if (c < HID) {
      const float a = (float)u[r * N1P + c];
      const float g = (float)u[r * N1P + HID + c];
      v = a * 0.5f * g * (1.0f + erff(g * 0.70710678f));
    }
    act[i] = (__bf16)v;
  }
}

// fp32 (KxN) -> bf16 transposed (NpxKp) with zero padding. B operands of all
// GEMMs are weights; transposing here makes LDS B-fragment loads contiguous.
__global__ __launch_bounds__(256) void f32_to_bf16_transpose_pad(
    const float* __restrict__ in, int kv, int nv,
    __bf16* __restrict__ out, int np, int kp) {
  const size_t total = (size_t)np * kp;
  for (size_t i = (size_t)blockIdx.x * blockDim.x + threadIdx.x; i < total;
       i += (size_t)gridDim.x * blockDim.x) {
    const size_t n = i / kp;
    const int k = (int)(i % kp);
    const float v = (n < (size_t)nv && k < kv) ? in[(size_t)k * nv + n] : 0.f;
    out[i] = (__bf16)v;
  }
}

// ---------------------------------------------------------------------------
extern "C" void kernel_launch(void* const* d_in, const int* in_sizes, int n_in,
                              void* d_out, int out_size, void* d_ws, size_t ws_size,
                              hipStream_t stream) {
  const float* x       = (const float*)d_in[0];
  const float* ln1_g   = (const float*)d_in[1];
  const float* ln1_b   = (const float*)d_in[2];
  const float* w_qkv   = (const float*)d_in[3];
  const float* q_scale = (const float*)d_in[4];
  const float* k_scale = (const float*)d_in[5];
  const float* w_out   = (const float*)d_in[6];
  const float* ln2_g   = (const float*)d_in[7];
  const float* ln2_b   = (const float*)d_in[8];
  const float* w1      = (const float*)d_in[9];
  const float* b1      = (const float*)d_in[10];
  const float* w2      = (const float*)d_in[11];
  const float* b2      = (const float*)d_in[12];
  float* out = (float*)d_out;

  char* p = (char*)d_ws;
  auto alloc = [&](size_t bytes) -> char* {
    char* r = p;
    p += (bytes + 255) & ~(size_t)255;
    return r;
  };
  __bf16* h_bf    = (__bf16*)alloc((size_t)SEQ * DIMX * 2);
  __bf16* wqkv_t  = (__bf16*)alloc((size_t)NQKV * DIMX * 2);    // 1536 x 1024
  __bf16* qkv_bf  = (__bf16*)alloc((size_t)SEQ * NQKV * 2);
  __bf16* q_bf    = (__bf16*)alloc((size_t)SEQ * INNER * 2);
  __bf16* k_bf    = (__bf16*)alloc((size_t)SEQ * INNER * 2);
  __bf16* vt_bf   = (__bf16*)alloc((size_t)SEQ * INNER * 2);    // [h][d][tok]
  __bf16* o_bf    = (__bf16*)alloc((size_t)SEQ * INNER * 2);
  __bf16* wout_t  = (__bf16*)alloc((size_t)DIMX * INNER * 2);   // 1024 x 512
  __bf16* hf_bf   = (__bf16*)alloc((size_t)SEQ * DIMX * 2);
  __bf16* w1_t    = (__bf16*)alloc((size_t)N1P * DIMX * 2);     // 5504 x 1024
  __bf16* u_bf    = (__bf16*)alloc((size_t)SEQ * N1P * 2);
  __bf16* act_bf  = (__bf16*)alloc((size_t)SEQ * HID_P * 2);
  __bf16* w2_t    = (__bf16*)alloc((size_t)DIMX * HID_P * 2);   // 1024 x 2752
  (void)ws_size; (void)in_sizes; (void)n_in; (void)out_size;

  const dim3 blk256(256);

  // Weight conversions: fp32 -> bf16, transposed to NxK, zero-padded.
  f32_to_bf16_transpose_pad<<<dim3(2048), blk256, 0, stream>>>(w_qkv, DIMX, NQKV,
                                                              wqkv_t, NQKV, DIMX);
  f32_to_bf16_transpose_pad<<<dim3(1024), blk256, 0, stream>>>(w_out, INNER, DIMX,
                                                              wout_t, DIMX, INNER);
  f32_to_bf16_transpose_pad<<<dim3(4096), blk256, 0, stream>>>(w1, DIMX, N1V,
                                                              w1_t, N1P, DIMX);
  f32_to_bf16_transpose_pad<<<dim3(2048), blk256, 0, stream>>>(w2, HID, DIMX,
                                                              w2_t, DIMX, HID_P);

  // LN1 -> bf16
  ln_bf16_kernel<<<dim3(SEQ), blk256, 0, stream>>>(x, ln1_g, ln1_b, h_bf);

  // QKV GEMM: (8192x1536) = h(8192x1024) * Wqkv, bf16 out.
  gemm_wmma<true><<<dim3(NQKV / 128, SEQ / 128), blk256, 0, stream>>>(
      h_bf, DIMX, wqkv_t, DIMX, qkv_bf, NQKV, nullptr, 0, nullptr, SEQ, NQKV, DIMX);

  // Per-head l2norm + scales + xpos rotary; v transposed to [h][d][tok].
  qkv_post_kernel<<<dim3(SEQ, HEADS), dim3(64), 0, stream>>>(qkv_bf, q_scale, k_scale,
                                                             q_bf, k_bf, vt_bf);

  // Local attention.
  attn_kernel<<<dim3(WSZ / 64, NWIN, HEADS), dim3(128), 0, stream>>>(q_bf, k_bf, vt_bf, o_bf);

  // Out-proj GEMM + residual(x) -> d_out (fp32 x1).
  gemm_wmma<false><<<dim3(DIMX / 128, SEQ / 128), blk256, 0, stream>>>(
      o_bf, INNER, wout_t, INNER, out, DIMX, nullptr, 0, x, SEQ, DIMX, INNER);

  // LN2 on x1 -> bf16
  ln_bf16_kernel<<<dim3(SEQ), blk256, 0, stream>>>(out, ln2_g, ln2_b, hf_bf);

  // FFN1 GEMM + b1 -> u (bf16, N padded 5460->5504).
  gemm_wmma<true><<<dim3(N1P / 128, SEQ / 128), blk256, 0, stream>>>(
      hf_bf, DIMX, w1_t, DIMX, u_bf, N1P, b1, N1V, nullptr, SEQ, N1P, DIMX);

  // GEGLU -> act (bf16, K padded 2730->2752 with zeros).
  geglu_kernel<<<dim3(4096), blk256, 0, stream>>>(u_bf, act_bf);

  // FFN2 GEMM + b2 + residual(x1 in d_out) -> d_out.
  gemm_wmma<false><<<dim3(DIMX / 128, SEQ / 128), blk256, 0, stream>>>(
      act_bf, HID_P, w2_t, HID_P, out, DIMX, b2, DIMX, out, SEQ, DIMX, HID_P);
}
